// RNNInfer_24644522344598
// MI455X (gfx1250) — matrix operand order repaired and checked by reading, not compiled, not run on hardware
//
#include <hip/hip_runtime.h>
#include <math.h>

// Problem constants (match reference)
#define B_ 32
#define T_ 1024
#define I_ 128
#define H_ 256

typedef __attribute__((ext_vector_type(16))) _Float16 v16h;
typedef __attribute__((ext_vector_type(8)))  _Float16 v8h;
typedef __attribute__((ext_vector_type(8)))  float    v8f;

// D = A(16x32 f16) * B(32x16 f16) + C(16x16 f32)
#define WMMA(a, b, c) \
  __builtin_amdgcn_wmma_f32_16x16x32_f16(false, (a), false, (b), (short)0, (c), false, false)

// ---------------------------------------------------------------------------
// Fragment loaders (wave32); A and B both use contiguous k = k0 + 16*(lane/16) + i.
static __device__ inline v16h ldA(const _Float16* __restrict__ base, int stride,
                                  int m0, int k0, int lane) {
  const int s = lane >> 4, m = lane & 15;
  const _Float16* p0 = base + (m0 + m) * stride + k0 + 16 * s;
  v16h r;
  *(v8h*)&r       = *(const v8h*)p0;        // K = k0+16s .. +7
  *((v8h*)&r + 1) = *(const v8h*)(p0 + 8);  // K = k0+16s+8 .. +7
  return r;
}
// B-matrix 32x16 (KxN): out = x @ W^T with row-major W[H][K] means
// B[k][n] = W[j0+n][k] -> one contiguous 32-byte row chunk per lane.
static __device__ inline v16h ldB(const _Float16* __restrict__ W, int stride,
                                  int j0, int k0, int lane) {
  const int s = lane >> 4, n = lane & 15;
  return *(const v16h*)(W + (j0 + n) * stride + k0 + 16 * s);
}
// Same B fragment from global f32 weights (once, to make fragments resident).
static __device__ inline v16h ldB_f32(const float* __restrict__ W, int stride,
                                      int j0, int k0, int lane) {
  const int s = lane >> 4, n = lane & 15;
  const float* p = W + (size_t)(j0 + n) * stride + k0 + 16 * s;
  v16h r;
#pragma unroll
  for (int i = 0; i < 16; ++i) r[i] = (_Float16)p[i];
  return r;
}

// Deterministic pseudo-noise. Reference adds NOISE_STD*DIAG*eps = 1e-6*eps,
// below fp16 activation resolution; exact threefry reproduction not needed.
static __device__ inline float cheap_noise(unsigned int x) {
  x ^= x >> 16; x *= 0x7feb352dU; x ^= x >> 15; x *= 0x846ca68bU; x ^= x >> 16;
  float u1 = (float)(x & 0xffffu) * (1.0f / 65536.0f);
  float u2 = (float)(x >> 16)     * (1.0f / 65536.0f);
  return (u1 + u2 - 1.0f) * 2.449f;
}

// LDS byte offset of a __shared__ object = low 32 bits of its flat address
// (LDS aperture keeps the offset in addr[31:0], CDNA5 ISA).
static __device__ inline unsigned lds_off(const void* p) {
  return (unsigned)(size_t)p;
}

// ---------------------------------------------------------------------------
// Kernel 1: y_train [B,T,I] f32  ->  x16 [T,B,I] f16 (contiguous per step)
__global__ __launch_bounds__(256) void cvt_x_kernel(const float* __restrict__ y,
                                                    _Float16* __restrict__ x16) {
  int g = (blockIdx.x * 256 + threadIdx.x) * 4;
  int t = g >> 12;                 // / (B*I = 4096)
  int r = g & 4095;
  int b = r >> 7, i = r & 127;
  const float4 v = *(const float4*)(y + ((size_t)b * T_ + t) * I_ + i);
  _Float16* d = x16 + g;
  d[0] = (_Float16)v.x; d[1] = (_Float16)v.y;
  d[2] = (_Float16)v.z; d[3] = (_Float16)v.w;
}

// ---------------------------------------------------------------------------
// Kernel 2: block 0 = forward RNN, block 1 = backward RNN. 16 waves; wave w
// owns output columns [16w,16w+16). Weight B-fragments register-resident.
// x_t is double-buffered in LDS and filled one step ahead with CDNA5
// GLOBAL_LOAD_ASYNC_TO_LDS_B128 (ASYNCcnt) so the global->LDS latency hides
// behind the 24 WMMAs of the current step.
__global__ __launch_bounds__(512) void rnn_kernel(
    const _Float16* __restrict__ x16,
    const float* __restrict__ Wf_ih, const float* __restrict__ Wf_hh,
    const float* __restrict__ bf_ih, const float* __restrict__ bf_hh,
    const float* __restrict__ Wb_ih, const float* __restrict__ Wb_hh,
    const float* __restrict__ bb_ih, const float* __restrict__ bb_hh,
    _Float16* __restrict__ fo16, _Float16* __restrict__ bo16) {
  __shared__ _Float16 xb[2][B_ * I_];       // 16 KB (ping-pong input)
  __shared__ _Float16 hbuf[2][B_ * H_];     // 32 KB (ping-pong hidden state)

  const int tid = threadIdx.x, lane = tid & 31, w = tid >> 5;
  const int bwd = blockIdx.x;
  const float* Wi = bwd ? Wb_ih : Wf_ih;
  const float* Wh = bwd ? Wb_hh : Wf_hh;
  const float* bi = bwd ? bb_ih : bf_ih;
  const float* bh = bwd ? bb_hh : bf_hh;
  _Float16* out = bwd ? bo16 : fo16;

  const int j0 = w * 16, n = lane & 15, sg = lane >> 4;
  const float bias = bi[j0 + n] + bh[j0 + n];

  v16h wif[4], whf[8];
#pragma unroll
  for (int k = 0; k < 4; ++k) wif[k] = ldB_f32(Wi, I_, j0, 32 * k, lane);
#pragma unroll
  for (int k = 0; k < 8; ++k) whf[k] = ldB_f32(Wh, H_, j0, 32 * k, lane);

  for (int idx = tid; idx < B_ * H_; idx += 512) hbuf[0][idx] = (_Float16)0.f;

  const unsigned ldsx[2] = {lds_off(&xb[0][0]) + (unsigned)tid * 16u,
                            lds_off(&xb[1][0]) + (unsigned)tid * 16u};

  // issue async stage of x for t=0 into buffer 0
  {
    const _Float16* gp = x16 + (size_t)(bwd ? T_ - 1 : 0) * (B_ * I_) + tid * 8;
    asm volatile("global_load_async_to_lds_b128 %0, %1, off"
                 :: "v"(ldsx[0]), "v"(gp) : "memory");
  }
  __syncthreads();

  int pb = 0;
  for (int t = 0; t < T_; ++t) {
    const int tt = bwd ? (T_ - 1 - t) : t;
    if (t + 1 < T_) {            // prefetch x_{t+1} into the other buffer
      const int tn = bwd ? (T_ - 2 - t) : (t + 1);
      const _Float16* gp = x16 + (size_t)tn * (B_ * I_) + tid * 8;
      asm volatile("global_load_async_to_lds_b128 %0, %1, off"
                   :: "v"(ldsx[(t + 1) & 1]), "v"(gp) : "memory");
      asm volatile("s_wait_asynccnt 0x1" ::: "memory");  // current buf done
    } else {
      asm volatile("s_wait_asynccnt 0x0" ::: "memory");
    }
    __syncthreads();

    const _Float16* xp = xb[t & 1];
    const _Float16* hp = hbuf[pb];
    v8f acc0 = {}, acc1 = {};
    // 12 K-slices (4 from x@W_ih^T, 8 from h@W_hh^T), software-pipelined so
    // each A-fragment pair is loaded one full slice (2 WMMAs) ahead of use.
    v16h a0 = ldA(xp, I_, 0, 0, lane);
    v16h a1 = ldA(xp, I_, 16, 0, lane);
#pragma unroll
    for (int k = 0; k < 12; ++k) {
      v16h n0 = a0, n1 = a1;
      if (k + 1 < 12) {
        const _Float16* src = (k + 1 < 4) ? xp : hp;
        const int str = (k + 1 < 4) ? I_ : H_;
        const int kk = (k + 1 < 4) ? 32 * (k + 1) : 32 * (k - 3);
        n0 = ldA(src, str, 0, kk, lane);
        n1 = ldA(src, str, 16, kk, lane);
      }
      const v16h bfrag = (k < 4) ? wif[k] : whf[k - 4];
      acc0 = WMMA(a0, bfrag, acc0);
      acc1 = WMMA(a1, bfrag, acc1);
      a0 = n0; a1 = n1;
    }

    _Float16* hn = hbuf[pb ^ 1];
    _Float16* og = out + (size_t)tt * (B_ * H_);
#pragma unroll
    for (int j = 0; j < 8; ++j) {           // C/D: M = j + 8*sg (+16 for acc1)
      int m0 = j + 8 * sg;
      float h0 = tanhf(acc0[j] + bias);
      float h1 = tanhf(acc1[j] + bias);
      hn[m0 * H_ + j0 + n]        = (_Float16)h0;
      hn[(m0 + 16) * H_ + j0 + n] = (_Float16)h1;
      og[m0 * H_ + j0 + n]        = (_Float16)h0;
      og[(m0 + 16) * H_ + j0 + n] = (_Float16)h1;
    }
    __syncthreads();   // xbuf/hbuf reads done before overwrite next iter
    pb ^= 1;
  }
}

// ---------------------------------------------------------------------------
// Kernel 3: sequential decoder. Wm_ih/Wm_hh/W2 fragments register-resident
// (144 VGPRs); W1 in LDS f16 (dynamic LDS = 116 KB; CDNA5 WGP has 320 KB).
// f_out/b_out tiles for step t+1 are prefetched into registers during the
// compute phases of step t.
__global__ __launch_bounds__(512) void dec_kernel(
    const _Float16* __restrict__ fo16, const _Float16* __restrict__ bo16,
    const float* __restrict__ Wm_ih, const float* __restrict__ Wm_hh,
    const float* __restrict__ bm_ih, const float* __restrict__ bm_hh,
    const float* __restrict__ W1, const float* __restrict__ b1,
    const float* __restrict__ W2, const float* __restrict__ b2,
    float* __restrict__ out_s, float* __restrict__ out_m) {
  extern __shared__ _Float16 smem[];
  _Float16* W1s = smem;                 // [64][512]  64 KB
  _Float16* emb = W1s + 64 * 512;       // [32][512]  32 KB (0..255 bi_t/x2; 256..511 s_prev)
  _Float16* z1  = emb + 32 * 512;       // [32][64]    4 KB
  _Float16* hb  = z1 + 32 * 64;         // [32][256]  16 KB (h_last = prev mean)

  const int tid = threadIdx.x, lane = tid & 31, w = tid >> 5;
  const int n = lane & 15, sg = lane >> 4, j0 = w * 16;

  for (int idx = tid; idx < 64 * 512; idx += 512) W1s[idx] = (_Float16)W1[idx];
  for (int idx = tid; idx < 32 * 256; idx += 512) {
    hb[idx] = (_Float16)0.f;
    emb[(idx >> 8) * 512 + 256 + (idx & 255)] = (_Float16)0.f;   // s_prev = 0
  }

  v16h wmih[8], wmhh[8], w2f[2];
#pragma unroll
  for (int k = 0; k < 8; ++k) wmih[k] = ldB_f32(Wm_ih, H_, j0, 32 * k, lane);
#pragma unroll
  for (int k = 0; k < 8; ++k) wmhh[k] = ldB_f32(Wm_hh, H_, j0, 32 * k, lane);
#pragma unroll
  for (int k = 0; k < 2; ++k) w2f[k] = ldB_f32(W2, 64, j0, 32 * k, lane);

  const float bm  = bm_ih[j0 + n] + bm_hh[j0 + n];
  const float b2c = b2[j0 + n];
  const float b1c = (w < 4) ? b1[w * 16 + n] : 0.f;

  // register prefetch of f_out/b_out tile for t = 0 (16 halves per thread)
  const int base = tid * 16;
  const int brow = base >> 8, bcol = base & 255;
  v8h f0 = *(const v8h*)(fo16 + base), f1 = *(const v8h*)(fo16 + base + 8);
  v8h g0 = *(const v8h*)(bo16 + base), g1 = *(const v8h*)(bo16 + base + 8);
  __syncthreads();

  for (int t = 0; t < T_; ++t) {
    { // phase 1: bi_t = f + b (from prefetched registers) -> emb cols 0..255
      _Float16* dp = emb + brow * 512 + bcol;
#pragma unroll
      for (int e = 0; e < 8; ++e) {
        dp[e]     = (_Float16)((float)f0[e] + (float)g0[e]);
        dp[e + 8] = (_Float16)((float)f1[e] + (float)g1[e]);
      }
    }
    __syncthreads();

    // issue global prefetch for step t+1; consumed next iteration, so the
    // L2 latency hides behind phases 2-5.
    v8h f0n = f0, f1n = f1, g0n = g0, g1n = g1;
    if (t + 1 < T_) {
      const _Float16* fp = fo16 + (size_t)(t + 1) * (B_ * H_) + base;
      const _Float16* bp = bo16 + (size_t)(t + 1) * (B_ * H_) + base;
      f0n = *(const v8h*)fp; f1n = *(const v8h*)(fp + 8);
      g0n = *(const v8h*)bp; g1n = *(const v8h*)(bp + 8);
    }

    if (w < 4) { // phase 2: z1 = relu(emb @ W1^T + b1)  [32,512]x[512,64]
      v8f a0 = {}, a1 = {};
#pragma unroll
      for (int k = 0; k < 16; ++k) {
        v16h bf = ldB(W1s, 512, j0, 32 * k, lane);
        v16h x0 = ldA(emb, 512, 0, 32 * k, lane);
        v16h x1 = ldA(emb, 512, 16, 32 * k, lane);
        a0 = WMMA(x0, bf, a0);
        a1 = WMMA(x1, bf, a1);
      }
#pragma unroll
      for (int j = 0; j < 8; ++j) {
        int m0 = j + 8 * sg;
        float v0 = a0[j] + b1c; v0 = v0 > 0.f ? v0 : 0.f;
        float v1 = a1[j] + b1c; v1 = v1 > 0.f ? v1 : 0.f;
        z1[m0 * 64 + j0 + n]        = (_Float16)v0;
        z1[(m0 + 16) * 64 + j0 + n] = (_Float16)v1;
      }
    }
    __syncthreads();

    { // phase 3: x2 = z1 @ W2^T + b2 -> overlay into emb cols 0..255
      v8f a0 = {}, a1 = {};
#pragma unroll
      for (int k = 0; k < 2; ++k) {
        v16h x0 = ldA(z1, 64, 0, 32 * k, lane);
        v16h x1 = ldA(z1, 64, 16, 32 * k, lane);
        a0 = WMMA(x0, w2f[k], a0);
        a1 = WMMA(x1, w2f[k], a1);
      }
#pragma unroll
      for (int j = 0; j < 8; ++j) {
        int m0 = j + 8 * sg;
        emb[m0 * 512 + j0 + n]        = (_Float16)(a0[j] + b2c);
        emb[(m0 + 16) * 512 + j0 + n] = (_Float16)(a1[j] + b2c);
      }
    }
    __syncthreads();

    // phase 4: pre = x2 @ Wm_ih^T + h_last @ Wm_hh^T   (2 x K=256)
    v8f a0 = {}, a1 = {};
#pragma unroll
    for (int k = 0; k < 8; ++k) {
      v16h x0 = ldA(emb, 512, 0, 32 * k, lane);  // x2 lives in emb cols 0..255
      v16h x1 = ldA(emb, 512, 16, 32 * k, lane);
      a0 = WMMA(x0, wmih[k], a0);
      a1 = WMMA(x1, wmih[k], a1);
    }
#pragma unroll
    for (int k = 0; k < 8; ++k) {
      v16h h0 = ldA(hb, 256, 0, 32 * k, lane);
      v16h h1 = ldA(hb, 256, 16, 32 * k, lane);
      a0 = WMMA(h0, wmhh[k], a0);
      a1 = WMMA(h1, wmhh[k], a1);
    }
    __syncthreads();   // all reads of hb/emb done before phase-5 overwrites

    // phase 5: mean = tanh(pre + b); sampled = mean + 1e-6*eps
#pragma unroll
    for (int j = 0; j < 8; ++j) {
      int m0 = j + 8 * sg;
#pragma unroll
      for (int r = 0; r < 2; ++r) {
        int row = m0 + 16 * r;
        float pre  = (r ? a1[j] : a0[j]) + bm;
        float mean = tanhf(pre);
        float eps  = cheap_noise((unsigned)t * 8192u +
                                 (unsigned)(row * 256 + j0 + n));
        float samp = mean + 1e-6f * eps;        // NOISE_STD * DIAG = 1e-6
        hb[row * H_ + j0 + n]         = (_Float16)mean;
        emb[row * 512 + 256 + j0 + n] = (_Float16)samp;  // next s_prev
        size_t gi = ((size_t)row * T_ + t) * H_ + j0 + n;  // [B,T,H]
        out_s[gi] = samp;
        out_m[gi] = mean;
      }
    }
    __syncthreads();
    f0 = f0n; f1 = f1n; g0 = g0n; g1 = g1n;
  }
}

// ---------------------------------------------------------------------------
extern "C" void kernel_launch(void* const* d_in, const int* in_sizes, int n_in,
                              void* d_out, int out_size, void* d_ws, size_t ws_size,
                              hipStream_t stream) {
  (void)in_sizes; (void)n_in; (void)out_size; (void)ws_size;
  const float* y     = (const float*)d_in[0];
  const float* Wf_ih = (const float*)d_in[1];
  const float* Wf_hh = (const float*)d_in[2];
  const float* bf_ih = (const float*)d_in[3];
  const float* bf_hh = (const float*)d_in[4];
  const float* Wb_ih = (const float*)d_in[5];
  const float* Wb_hh = (const float*)d_in[6];
  const float* bb_ih = (const float*)d_in[7];
  const float* bb_hh = (const float*)d_in[8];
  const float* Wm_ih = (const float*)d_in[9];
  const float* Wm_hh = (const float*)d_in[10];
  const float* bm_ih = (const float*)d_in[11];
  const float* bm_hh = (const float*)d_in[12];
  const float* W1    = (const float*)d_in[13];
  const float* b1    = (const float*)d_in[14];
  const float* W2    = (const float*)d_in[15];
  const float* b2    = (const float*)d_in[16];

  char* ws = (char*)d_ws;
  _Float16* x16  = (_Float16*)ws;                                   // 8 MB
  _Float16* fo16 = (_Float16*)(ws + (size_t)T_ * B_ * I_ * 2);      // 16 MB
  _Float16* bo16 = fo16 + (size_t)T_ * B_ * H_;                     // 16 MB
  float* out_s = (float*)d_out;
  float* out_m = out_s + (size_t)B_ * T_ * H_;

  cvt_x_kernel<<<(B_ * T_ * I_) / (256 * 4), 256, 0, stream>>>(y, x16);
  rnn_kernel<<<2, 512, 0, stream>>>(x16, Wf_ih, Wf_hh, bf_ih, bf_hh,
                                    Wb_ih, Wb_hh, bb_ih, bb_hh, fo16, bo16);

  const int smem = (64 * 512 + 32 * 512 + 32 * 64 + 32 * 256) * (int)sizeof(_Float16);
  (void)hipFuncSetAttribute((const void*)dec_kernel,
                            hipFuncAttributeMaxDynamicSharedMemorySize, smem);
  dec_kernel<<<1, 512, smem, stream>>>(fo16, bo16, Wm_ih, Wm_hh, bm_ih, bm_hh,
                                       W1, b1, W2, b2, out_s, out_m);
}
